// RawFormer_82729660055625
// MI455X (gfx1250) — matrix operand order repaired and checked by reading, not compiled, not run on hardware
//
#include <hip/hip_runtime.h>
#include <math.h>

// ---------------- types / WMMA helpers (wave32, gfx1250) ----------------
typedef __attribute__((ext_vector_type(16))) _Float16 v16h;
typedef __attribute__((ext_vector_type(8)))  _Float16 v8h;
typedef __attribute__((ext_vector_type(8)))  float    v8f;

union V16 { v16h v; v8h p[2]; _Float16 h[16]; };

__device__ __forceinline__ v8f wmma16(v16h a, v16h b, v8f c) {
  // D = A(16x32 f16) * B(32x16 f16) + C(16x16 f32)
  return __builtin_amdgcn_wmma_f32_16x16x32_f16(false, a, false, b, (short)0, c, false, false);
}

// A-matrix 16x32 f16 packing, contiguous k (not the ISA table order): given row M (0..15) and k-in-chunk kk
// (0..31), return the (lane*16 + halfIdx) slot inside a 512-half packed tile.
__device__ __forceinline__ int amap_slot(int M, int kk) {
  int hi, q;
  hi = kk >> 4; q = kk & 15;
  return (M + hi * 16) * 16 + q;
}

// ---------------- constants ----------------
#define BATCH   4
#define LENGTH  65494
#define LPAD    65600
#define FILTS   256
#define KERNEL  1025
#define KCHUNKS 33          // ceil(1025/32)
#define CONVOUT 64470
#define TPOOL   21490
#define POSBLK  336         // 21490 pooled / (4 waves * 16) rounded up
#define NWIN    1256        // window halves per WG (192 + 1056 + pad)
#define DM      256
#define EMBCH   672         // ceil(21490/32)
#define DFF     1024

// ---------------- pack kernels ----------------
__global__ void k_pack_x(const float* __restrict__ x, _Float16* __restrict__ xh) {
  int i = blockIdx.x * 256 + threadIdx.x;          // BATCH*LPAD
  int b = i / LPAD, j = i - b * LPAD;
  float v = (j < LENGTH) ? x[b * LENGTH + j] : 0.f;
  xh[i] = (_Float16)v;
}

__global__ void k_pack_filt(const float* __restrict__ f, _Float16* __restrict__ fa) {
  int idx = blockIdx.x * 256 + threadIdx.x;        // 16*33*512
  int mt = idx / (KCHUNKS * 512); int r = idx - mt * (KCHUNKS * 512);
  int c = r / 512; int li = r - c * 512;
  int lane = li / 16, q = li - lane * 16;
  int hi = (lane >= 16) ? 1 : 0;
  int M = mt * 16 + (lane & 15);
  int k = c * 32 + q + hi * 16; // inverse of amap_slot
  float v = (k < KERNEL) ? f[M * KERNEL + k] : 0.f;
  fa[idx] = (_Float16)v;
}

// B-matrix 32x16 f16 layout: lane L holds n = L%16, K = q + (L<16?0:16) (contiguous).
__global__ void k_pack_w(const float* __restrict__ src, _Float16* __restrict__ dst,
                         int nTileOff, int kC, int Ksrc, int total) {
  int idx = blockIdx.x * 256 + threadIdx.x;
  if (idx >= total) return;
  int nt = idx / (kC * 512); int r = idx - nt * (kC * 512);
  int c = r / 512; int li = r - c * 512;
  int lane = li / 16, q = li - lane * 16;
  int n = nt * 16 + (lane & 15);
  int k = c * 32 + q + ((lane < 16) ? 0 : 16);
  float v = (k < Ksrc) ? src[(size_t)n * Ksrc + k] : 0.f;
  dst[((size_t)(nTileOff + nt) * kC + c) * 512 + li] = (_Float16)v;
}

__global__ void k_qkv_bias(const float* bq, const float* bk, const float* bv, float* dst) {
  int idx = blockIdx.x * 256 + threadIdx.x;        // 4*768
  int l = idx / 768, j = idx - l * 768;
  float v = (j < 256) ? bq[l * 256 + j] : (j < 512) ? bk[l * 256 + j - 256]
                                                    : bv[l * 256 + j - 512];
  dst[idx] = v;
}

// ---------------- SincConv as WMMA GEMM + |x| + maxpool3 + BN stats ----------------
__global__ void __launch_bounds__(128)
k_sincconv(const _Float16* __restrict__ xh, const _Float16* __restrict__ fa,
           float* __restrict__ hbuf, float* __restrict__ bn1sum) {
  const int tid = threadIdx.x, lane = tid & 31, wave = tid >> 5;
  const int blk = blockIdx.x, mTile = blockIdx.y, b = blockIdx.z;

  __shared__ __align__(32) _Float16 ldsWin[8 * NWIN];   // 8 phase-shifted copies
  __shared__ float ldsConv[4][16][48];
  __shared__ float ldsRed[32];
  if (tid < 32) ldsRed[tid] = 0.f;

  const int p0 = blk * 192;                             // conv base position of WG
  const _Float16* xb = xh + (size_t)b * LPAD + p0;
  for (int i = tid; i < 8 * NWIN; i += 128) {
    int ph = i / NWIN, j = i - ph * NWIN;
    ldsWin[i] = xb[j + ph];                             // C_p[j] = win[j+p]
  }
  __syncthreads();

  v8f acc0 = {}, acc1 = {}, acc2 = {};
  const _Float16* fat = fa + (size_t)mTile * KCHUNKS * 512 + lane * 16;
  const int nlo = lane & 15;
  const int koff = (lane < 16) ? 0 : 16;
  const int pw = wave * 48;

  for (int c = 0; c < KCHUNKS; ++c) {
    V16 a;
    a.p[0] = *(const v8h*)(fat + c * 512);
    a.p[1] = *(const v8h*)(fat + c * 512 + 8);
    int sBase = pw + nlo + c * 32 + koff;
    {
      int s = sBase; int ph = s & 7;
      const _Float16* wp = ldsWin + ph * NWIN + (s - ph);
      V16 bb; bb.p[0] = *(const v8h*)wp; bb.p[1] = *(const v8h*)(wp + 8);
      acc0 = wmma16(a.v, bb.v, acc0);
    }
    {
      int s = sBase + 16; int ph = s & 7;
      const _Float16* wp = ldsWin + ph * NWIN + (s - ph);
      V16 bb; bb.p[0] = *(const v8h*)wp; bb.p[1] = *(const v8h*)(wp + 8);
      acc1 = wmma16(a.v, bb.v, acc1);
    }
    {
      int s = sBase + 32; int ph = s & 7;
      const _Float16* wp = ldsWin + ph * NWIN + (s - ph);
      V16 bb; bb.p[0] = *(const v8h*)wp; bb.p[1] = *(const v8h*)(wp + 8);
      acc2 = wmma16(a.v, bb.v, acc2);
    }
  }

  const int mhi = (lane < 16) ? 0 : 8;                  // D-layout row offset
#pragma unroll
  for (int r = 0; r < 8; ++r) {
    int m = r + mhi;
    ldsConv[wave][m][nlo]      = fabsf(acc0[r]);
    ldsConv[wave][m][16 + nlo] = fabsf(acc1[r]);
    ldsConv[wave][m][32 + nlo] = fabsf(acc2[r]);
  }
  __syncthreads();

  const int tg = blk * 64 + wave * 16 + nlo;            // pooled index
#pragma unroll
  for (int r = 0; r < 8; ++r) {
    int m = r + mhi;
    float p = fmaxf(fmaxf(ldsConv[wave][m][3 * nlo], ldsConv[wave][m][3 * nlo + 1]),
                    ldsConv[wave][m][3 * nlo + 2]);
    if (tg < TPOOL) {
      hbuf[(size_t)(b * FILTS + mTile * 16 + m) * TPOOL + tg] = p;
      atomicAdd(&ldsRed[m], p);
      atomicAdd(&ldsRed[16 + m], p * p);
    }
  }
  __syncthreads();
  if (tid < 16) {
    atomicAdd(&bn1sum[mTile * 16 + tid], ldsRed[tid]);
    atomicAdd(&bn1sum[256 + mTile * 16 + tid], ldsRed[16 + tid]);
  }
}

__global__ void k_bn1_fin(const float* sums, const float* g, const float* bb, float* sc) {
  int c = threadIdx.x;
  const float N = (float)(BATCH * TPOOL);
  float m = sums[c] / N;
  float v = sums[256 + c] / N - m * m;
  float s = g[c] * rsqrtf(v + 1e-5f);
  sc[c] = s;
  sc[256 + c] = bb[c] - m * s;
}

// ------------- embedding GEMM: e = selu(bn(h)) @ emb_w^T, K=21490 -------------
__global__ void __launch_bounds__(512)
k_emb_gemm(const float* __restrict__ hbuf, const float* __restrict__ bn1sc,
           const _Float16* __restrict__ embB, const float* __restrict__ emb_b,
           float* __restrict__ eraw, float* __restrict__ bn2sum) {
  const int tid = threadIdx.x, lane = tid & 31, wave = tid >> 5;
  const int cTile = blockIdx.x, b = blockIdx.y;
  const int nTile = wave;                               // 16 waves -> 16 d-tiles
  __shared__ __align__(32) _Float16 ldsA[2][512];
  const int c_r = tid >> 5, tq = tid & 31;
  const int slot = amap_slot(c_r, tq);
  const float scl = bn1sc[cTile * 16 + c_r];
  const float sh  = bn1sc[256 + cTile * 16 + c_r];
  const float* hrow = hbuf + (size_t)(b * FILTS + cTile * 16 + c_r) * TPOOL;

  v8f acc = {};
  const _Float16* wB = embB + (size_t)nTile * EMBCH * 512 + lane * 16;
  for (int ch = 0; ch < EMBCH; ++ch) {
    int t = ch * 32 + tq;
    float av = 0.f;
    if (t < TPOOL) {
      float y = hrow[t] * scl + sh;
      av = 1.0507009873554805f * (y > 0.f ? y : 1.6732632423543772f * (expf(y) - 1.f));
    }
    ldsA[ch & 1][slot] = (_Float16)av;
    __syncthreads();
    V16 a; a.p[0] = *(const v8h*)&ldsA[ch & 1][lane * 16];
           a.p[1] = *(const v8h*)&ldsA[ch & 1][lane * 16 + 8];
    V16 bb; bb.p[0] = *(const v8h*)(wB + ch * 512);
            bb.p[1] = *(const v8h*)(wB + ch * 512 + 8);
    int chp = (ch + 4 < EMBCH) ? ch + 4 : ch;
    __builtin_prefetch((const void*)(wB + chp * 512), 0, 1);  // global_prefetch_b8
    acc = wmma16(a.v, bb.v, acc);
  }
  int n = nTile * 16 + (lane & 15);
  float bias = emb_b[n];
  int mhi = (lane < 16) ? 0 : 8;
  float ps = 0.f, pq = 0.f;
#pragma unroll
  for (int r = 0; r < 8; ++r) {
    int c = cTile * 16 + r + mhi;
    float e = acc[r] + bias;
    eraw[(size_t)(b * FILTS + c) * DM + n] = e;
    ps += e; pq += e * e;
  }
  atomicAdd(&bn2sum[n], ps);
  atomicAdd(&bn2sum[256 + n], pq);
}

__global__ void k_bn2_pos(const float* __restrict__ eraw, const float* __restrict__ sums,
                          const float* __restrict__ g, const float* __restrict__ bb,
                          float* __restrict__ z) {
  int idx = blockIdx.x * 256 + threadIdx.x;             // 4*256*256
  int d = idx & 255, c = (idx >> 8) & 255;
  const float N = (float)(BATCH * FILTS);
  float m = sums[d] / N;
  float v = sums[256 + d] / N - m * m;
  float y = (eraw[idx] - m) * rsqrtf(v + 1e-5f) * g[d] + bb[d];
  y = fmaxf(y, 0.f);
  float div = expf((float)(d & ~1) * (-9.210340371976184f / 256.f));
  float ang = (float)c * div;
  z[idx] = y + ((d & 1) ? cosf(ang) : sinf(ang));
}

// ------------- generic WMMA GEMM: out = act(A @ W^T + bias [+ resid]) -------------
__global__ void __launch_bounds__(512)
k_gemm(const float* __restrict__ A, const _Float16* __restrict__ W,
       const float* __restrict__ bias, const float* __restrict__ resid,
       float* __restrict__ outp, int K, int N, int act) {
  const int tid = threadIdx.x, lane = tid & 31, wave = tid >> 5;
  const int mTile = blockIdx.x, b = blockIdx.z;
  const int nTile = blockIdx.y * 16 + wave;
  const int kC = K >> 5;
  __shared__ __align__(32) _Float16 ldsA[2][512];
  const int c_r = tid >> 5, tq = tid & 31;
  const int slot = amap_slot(c_r, tq);
  const float* Ar = A + (size_t)(b * 256 + mTile * 16 + c_r) * K;

  v8f acc = {};
  const _Float16* wB = W + (size_t)nTile * kC * 512 + lane * 16;
  for (int ch = 0; ch < kC; ++ch) {
    ldsA[ch & 1][slot] = (_Float16)Ar[ch * 32 + tq];
    __syncthreads();
    V16 a; a.p[0] = *(const v8h*)&ldsA[ch & 1][lane * 16];
           a.p[1] = *(const v8h*)&ldsA[ch & 1][lane * 16 + 8];
    V16 bb; bb.p[0] = *(const v8h*)(wB + ch * 512);
            bb.p[1] = *(const v8h*)(wB + ch * 512 + 8);
    int chp = (ch + 2 < kC) ? ch + 2 : ch;
    __builtin_prefetch((const void*)(wB + chp * 512), 0, 1);
    acc = wmma16(a.v, bb.v, acc);
  }
  int n = nTile * 16 + (lane & 15);
  float bv = bias[n];
  int mhi = (lane < 16) ? 0 : 8;
#pragma unroll
  for (int r = 0; r < 8; ++r) {
    int m = mTile * 16 + r + mhi;
    float v = acc[r] + bv;
    if (resid) v += resid[(size_t)(b * 256 + m) * N + n];
    if (act == 1) v = fmaxf(v, 0.f);
    else if (act == 2) v = 0.5f * v * (1.f + erff(v * 0.70710678118654752f)); // exact GELU
    outp[(size_t)(b * 256 + m) * N + n] = v;
  }
}

// ------------- attention: per (b, head), Q/K/V packed in LDS, WMMA scores + PV -------------
__global__ void __launch_bounds__(128)
k_attn(const float* __restrict__ qkv, float* __restrict__ attnOut) {
  const int tid = threadIdx.x, lane = tid & 31, wave = tid >> 5;   // 4 waves
  const int h = blockIdx.x, b = blockIdx.y;
  extern __shared__ char smemRaw[];
  _Float16* sQ = (_Float16*)smemRaw;              // 16 sTiles * 512 halves (A-layout)
  _Float16* sK = sQ + 8192;                       // 16 s'Tiles * 512 (B-layout, K=hd)
  _Float16* sV = sK + 8192;                       // 16 chunks * 2 nTiles * 512 (B-layout)
  float* sS   = (float*)(sV + 16384);             // 4 waves * 16 rows * 256 cols
  float* sInv = sS + 4 * 16 * 256;                // 4*16

  const float* qb = qkv + (size_t)b * 256 * 768;
  for (int i = tid; i < 8192; i += 128) {
    int s = i >> 5, d = i & 31;
    int sT = s >> 4, M = s & 15;
    sQ[sT * 512 + amap_slot(M, d)] = (_Float16)qb[s * 768 + h * 32 + d];
    sK[sT * 512 + (M + ((d >= 16) ? 16 : 0)) * 16 + (d & 15)] =
        (_Float16)qb[s * 768 + 256 + h * 32 + d];
    int chv = s >> 5, kk = s & 31;
    sV[(chv * 2 + (d >> 4)) * 512 + ((d & 15) + ((kk >= 16) ? 16 : 0)) * 16 + (kk & 15)] =
        (_Float16)qb[s * 768 + 512 + h * 32 + d];
  }
  __syncthreads();

  const float scale = 0.17677669529663687f;       // 1/sqrt(32)
  float* sSw = sS + wave * 16 * 256;
  const int row = lane & 15;
  const int mhi = (lane < 16) ? 0 : 8;

  for (int it = 0; it < 4; ++it) {
    int sTile = it * 4 + wave;
    V16 qa; qa.p[0] = *(const v8h*)&sQ[sTile * 512 + lane * 16];
            qa.p[1] = *(const v8h*)&sQ[sTile * 512 + lane * 16 + 8];
    v8f accS[16];
#pragma unroll
    for (int j = 0; j < 16; ++j) {
      V16 kb; kb.p[0] = *(const v8h*)&sK[j * 512 + lane * 16];
              kb.p[1] = *(const v8h*)&sK[j * 512 + lane * 16 + 8];
      v8f zc = {};
      accS[j] = wmma16(qa.v, kb.v, zc);
    }
#pragma unroll
    for (int j = 0; j < 16; ++j)
#pragma unroll
      for (int r = 0; r < 8; ++r)
        sSw[(r + mhi) * 256 + j * 16 + row] = accS[j][r];
    __syncthreads();

    // softmax: lane pair (L, L+16) shares row L%16, each handles 128 cols
    int colBase = (lane < 16) ? 0 : 128;
    float mx = -3.4e38f;
    for (int cI = 0; cI < 128; ++cI) mx = fmaxf(mx, sSw[row * 256 + colBase + cI]);
    mx = fmaxf(mx, __shfl_xor(mx, 16));
    float sum = 0.f;
    for (int cI = 0; cI < 128; ++cI) {
      float e = expf(scale * (sSw[row * 256 + colBase + cI] - mx));
      sSw[row * 256 + colBase + cI] = e;
      sum += e;
    }
    sum += __shfl_xor(sum, 16);
    if (lane < 16) sInv[wave * 16 + lane] = 1.f / sum;
    __syncthreads();

    float rinv = sInv[wave * 16 + row];
    v8f accV0 = {}, accV1 = {};
    for (int ch = 0; ch < 16; ++ch) {
      V16 pa;
#pragma unroll
      for (int q = 0; q < 16; ++q) {
        int kk = q + ((lane >= 16) ? 16 : 0);
        pa.h[q] = (_Float16)(sSw[row * 256 + ch * 32 + kk] * rinv);
      }
      V16 vb0; vb0.p[0] = *(const v8h*)&sV[(ch * 2) * 512 + lane * 16];
               vb0.p[1] = *(const v8h*)&sV[(ch * 2) * 512 + lane * 16 + 8];
      accV0 = wmma16(pa.v, vb0.v, accV0);
      V16 vb1; vb1.p[0] = *(const v8h*)&sV[(ch * 2 + 1) * 512 + lane * 16];
               vb1.p[1] = *(const v8h*)&sV[(ch * 2 + 1) * 512 + lane * 16 + 8];
      accV1 = wmma16(pa.v, vb1.v, accV1);
    }
    float* ao = attnOut + (size_t)b * 256 * 256;
#pragma unroll
    for (int r = 0; r < 8; ++r) {
      int s = sTile * 16 + r + mhi;
      ao[s * 256 + h * 32 + row] = accV0[r];
      ao[s * 256 + h * 32 + 16 + row] = accV1[r];
    }
    __syncthreads();
  }
}

// ------------- LayerNorm over d=256 (one wave per row) -------------
__global__ void __launch_bounds__(256)
k_ln(const float* __restrict__ inp, const float* __restrict__ g,
     const float* __restrict__ bb, float* __restrict__ outp) {
  int wave = threadIdx.x >> 5, lane = threadIdx.x & 31;
  int rowg = blockIdx.x * 8 + wave;                     // 0..1023
  const float* r = inp + (size_t)rowg * 256;
  float v[8]; float s = 0.f;
#pragma unroll
  for (int i = 0; i < 8; ++i) { v[i] = r[lane * 8 + i]; s += v[i]; }
#pragma unroll
  for (int off = 16; off; off >>= 1) s += __shfl_xor(s, off);
  float mean = s * (1.f / 256.f);
  float sq = 0.f;
#pragma unroll
  for (int i = 0; i < 8; ++i) { float d = v[i] - mean; sq += d * d; }
#pragma unroll
  for (int off = 16; off; off >>= 1) sq += __shfl_xor(sq, off);
  float inv = rsqrtf(sq * (1.f / 256.f) + 1e-5f);
  float* o = outp + (size_t)rowg * 256;
#pragma unroll
  for (int i = 0; i < 8; ++i) {
    int d = lane * 8 + i;
    o[d] = (v[i] - mean) * inv * g[d] + bb[d];
  }
}

// ------------- mean-pool over seq then 256 -> 2 projection -------------
__global__ void k_pool_proj(const float* __restrict__ zf, const float* __restrict__ pw,
                            const float* __restrict__ pb, float* __restrict__ out) {
  int b = blockIdx.x, d = threadIdx.x;                  // 256 threads
  float s = 0.f;
  for (int i = 0; i < 256; ++i) s += zf[(size_t)(b * 256 + i) * 256 + d];
  float m = s * (1.f / 256.f);
  __shared__ float acc[2];
  if (d < 2) acc[d] = 0.f;
  __syncthreads();
  float p0 = m * pw[d], p1 = m * pw[256 + d];
#pragma unroll
  for (int off = 16; off; off >>= 1) { p0 += __shfl_xor(p0, off); p1 += __shfl_xor(p1, off); }
  if ((d & 31) == 0) { atomicAdd(&acc[0], p0); atomicAdd(&acc[1], p1); }
  __syncthreads();
  if (d < 2) out[b * 2 + d] = acc[d] + pb[d];
}

// ---------------- host: workspace layout + launch sequence ----------------
extern "C" void kernel_launch(void* const* d_in, const int* in_sizes, int n_in,
                              void* d_out, int out_size, void* d_ws, size_t ws_size,
                              hipStream_t stream) {
  (void)in_sizes; (void)n_in; (void)out_size; (void)ws_size;
  const float* x      = (const float*)d_in[0];
  const float* filt   = (const float*)d_in[1];
  const float* bn1_g  = (const float*)d_in[2];
  const float* bn1_b  = (const float*)d_in[3];
  const float* emb_w  = (const float*)d_in[4];
  const float* emb_b  = (const float*)d_in[5];
  const float* ebn_g  = (const float*)d_in[6];
  const float* ebn_b  = (const float*)d_in[7];
  const float* wq     = (const float*)d_in[8];
  const float* bq     = (const float*)d_in[9];
  const float* wk     = (const float*)d_in[10];
  const float* bk     = (const float*)d_in[11];
  const float* wv     = (const float*)d_in[12];
  const float* bv     = (const float*)d_in[13];
  const float* wo     = (const float*)d_in[14];
  const float* bo     = (const float*)d_in[15];
  const float* c1w    = (const float*)d_in[16];
  const float* c1b    = (const float*)d_in[17];
  const float* c2w    = (const float*)d_in[18];
  const float* c2b    = (const float*)d_in[19];
  const float* ln1g   = (const float*)d_in[20];
  const float* ln1b   = (const float*)d_in[21];
  const float* ln2g   = (const float*)d_in[22];
  const float* ln2b   = (const float*)d_in[23];
  const float* lnfg   = (const float*)d_in[24];
  const float* lnfb   = (const float*)d_in[25];
  const float* projw  = (const float*)d_in[26];
  const float* projb  = (const float*)d_in[27];

  char* ws = (char*)d_ws;
  size_t off = 0;
  auto AL = [&](size_t bytes) { size_t o = off; off += (bytes + 255) & ~(size_t)255; return o; };
  const size_t o_xh   = AL((size_t)BATCH * LPAD * 2);
  const size_t o_fa   = AL((size_t)16 * KCHUNKS * 512 * 2);
  const size_t o_h    = AL((size_t)BATCH * FILTS * TPOOL * 4);
  const size_t o_bn1s = AL(512 * 4);                    // conv BN sums (zeroed)
  const size_t o_bn2s = AL(512 * 4);                    // emb  BN sums (zeroed; adjacent)
  const size_t o_bn1c = AL(512 * 4);
  const size_t o_embB = AL((size_t)16 * EMBCH * 512 * 2);
  const size_t o_eraw = AL((size_t)BATCH * 256 * 256 * 4);
  const size_t o_z    = AL((size_t)BATCH * 256 * 256 * 4);
  const size_t o_qb   = AL((size_t)4 * 768 * 4);
  const size_t o_qkvW = AL((size_t)4 * 48 * 8 * 512 * 2);
  const size_t o_woW  = AL((size_t)4 * 16 * 8 * 512 * 2);
  const size_t o_c1W  = AL((size_t)4 * 64 * 8 * 512 * 2);
  const size_t o_c2W  = AL((size_t)4 * 16 * 32 * 512 * 2);
  const size_t o_qkv  = AL((size_t)BATCH * 256 * 768 * 4);
  const size_t o_attn = AL((size_t)BATCH * 256 * 256 * 4);
  const size_t o_pre  = AL((size_t)BATCH * 256 * 256 * 4);
  const size_t o_y    = AL((size_t)BATCH * 256 * DFF * 4);

  _Float16* xh   = (_Float16*)(ws + o_xh);
  _Float16* fa   = (_Float16*)(ws + o_fa);
  float*    hbuf = (float*)(ws + o_h);
  float*    bn1s = (float*)(ws + o_bn1s);
  float*    bn2s = (float*)(ws + o_bn2s);
  float*    bn1c = (float*)(ws + o_bn1c);
  _Float16* embB = (_Float16*)(ws + o_embB);
  float*    eraw = (float*)(ws + o_eraw);
  float*    z    = (float*)(ws + o_z);
  float*    qkvb = (float*)(ws + o_qb);
  _Float16* qkvW = (_Float16*)(ws + o_qkvW);
  _Float16* woW  = (_Float16*)(ws + o_woW);
  _Float16* c1W  = (_Float16*)(ws + o_c1W);
  _Float16* c2W  = (_Float16*)(ws + o_c2W);
  float*    qkvO = (float*)(ws + o_qkv);
  float*    attn = (float*)(ws + o_attn);
  float*    pre  = (float*)(ws + o_pre);
  float*    ybuf = (float*)(ws + o_y);

  hipMemsetAsync(ws + o_bn1s, 0, 4096, stream);         // bn1s + bn2s contiguous

  // --- packing (deterministic each call) ---
  k_pack_x<<<(BATCH * LPAD) / 256, 256, 0, stream>>>(x, xh);
  k_pack_filt<<<(16 * KCHUNKS * 512) / 256, 256, 0, stream>>>(filt, fa);
  k_pack_w<<<(16 * EMBCH * 512) / 256, 256, 0, stream>>>(emb_w, embB, 0, EMBCH, TPOOL,
                                                         16 * EMBCH * 512);
  k_qkv_bias<<<12, 256, 0, stream>>>(bq, bk, bv, qkvb);
  for (int l = 0; l < 4; ++l) {
    k_pack_w<<<256, 256, 0, stream>>>(wq + l * 65536, qkvW + (size_t)l * 196608, 0, 8, 256, 65536);
    k_pack_w<<<256, 256, 0, stream>>>(wk + l * 65536, qkvW + (size_t)l * 196608, 16, 8, 256, 65536);
    k_pack_w<<<256, 256, 0, stream>>>(wv + l * 65536, qkvW + (size_t)l * 196608, 32, 8, 256, 65536);
    k_pack_w<<<256, 256, 0, stream>>>(wo + l * 65536, woW + (size_t)l * 65536, 0, 8, 256, 65536);
    k_pack_w<<<1024, 256, 0, stream>>>(c1w + (size_t)l * 262144, c1W + (size_t)l * 262144,
                                       0, 8, 256, 262144);
    k_pack_w<<<1024, 256, 0, stream>>>(c2w + (size_t)l * 262144, c2W + (size_t)l * 262144,
                                       0, 32, 1024, 262144);
  }

  // --- front end ---
  k_sincconv<<<dim3(POSBLK, 16, BATCH), 128, 0, stream>>>(xh, fa, hbuf, bn1s);
  k_bn1_fin<<<1, 256, 0, stream>>>(bn1s, bn1_g, bn1_b, bn1c);
  k_emb_gemm<<<dim3(16, BATCH), 512, 0, stream>>>(hbuf, bn1c, embB, emb_b, eraw, bn2s);
  k_bn2_pos<<<(BATCH * 256 * 256) / 256, 256, 0, stream>>>(eraw, bn2s, ebn_g, ebn_b, z);

  // --- encoder layers ---
  const size_t ATTN_SMEM = (8192 + 8192 + 16384) * sizeof(_Float16) +
                           (4 * 16 * 256 + 4 * 16) * sizeof(float);
  for (int l = 0; l < 4; ++l) {
    k_gemm<<<dim3(16, 3, BATCH), 512, 0, stream>>>(z, qkvW + (size_t)l * 196608,
                                                   qkvb + l * 768, nullptr, qkvO, 256, 768, 0);
    k_attn<<<dim3(8, BATCH), 128, ATTN_SMEM, stream>>>(qkvO, attn);
    k_gemm<<<dim3(16, 1, BATCH), 512, 0, stream>>>(attn, woW + (size_t)l * 65536,
                                                   bo + l * 256, z, pre, 256, 256, 0);
    k_ln<<<128, 256, 0, stream>>>(pre, ln1g + l * 256, ln1b + l * 256, z);
    k_gemm<<<dim3(16, 4, BATCH), 512, 0, stream>>>(z, c1W + (size_t)l * 262144,
                                                   c1b + l * 1024, nullptr, ybuf, 256, 1024, 2);
    k_gemm<<<dim3(16, 1, BATCH), 512, 0, stream>>>(ybuf, c2W + (size_t)l * 262144,
                                                   c2b + l * 256, z, pre, 1024, 256, 0);
    k_ln<<<128, 256, 0, stream>>>(pre, ln2g + l * 256, ln2b + l * 256, z);
  }

  // --- head ---
  k_ln<<<128, 256, 0, stream>>>(z, lnfg, lnfb, pre);
  k_pool_proj<<<BATCH, 256, 0, stream>>>(pre, projw, projb, (float*)d_out);
}
